// LDGroupedAutoencoder_56281251447260
// MI455X (gfx1250) — compile-verified, matrix operand order checked by reading, not run on hardware
//
#include <hip/hip_runtime.h>
#include <hip/hip_bf16.h>

// ---------------------------------------------------------------------------
// LDGroupedAutoencoder on gfx1250:
//   1) enc grouped conv (VALU, bw-bound)  -> h bf16 [128,30000]
//   2) enc FC via bf16 WMMA (async A-tile loads), 8-way K-split -> partials
//   3) reduce + bias + leaky -> z bf16 [128,2000]
//   4) dec FC via bf16 WMMA, fused bias+leaky -> d f32 [128,30000]
//   5) dec grouped convT + sigmoid (VALU) -> out f32 [128,200000]
// ---------------------------------------------------------------------------

#define BATCH  128
#define NGRP   10000
#define KW     20
#define NCH    3
#define GC     30000     // NGRP*NCH
#define LAT    2000
#define INLEN  200000
#define NSPLIT 8
#define NPAD   2048

#define TM 128
#define TN 64
#define KT 32
#define SSTR 40          // LDS row stride in halfwords (80B) to spread banks

typedef __attribute__((ext_vector_type(16))) __bf16 v16bf;
typedef __attribute__((ext_vector_type(8)))  float  v8f;

union Frag16 { v16bf v; uint4 q[2]; };

// Round-to-nearest-even bf16 (used in the small elementwise kernels).
__device__ __forceinline__ unsigned short f2bf(float f) {
  unsigned int u = __float_as_uint(f);
  u += 0x7FFFu + ((u >> 16) & 1u);
  return (unsigned short)(u >> 16);
}
// Fast pack for weight staging: round-half-up + single v_perm_b32 merge.
__device__ __forceinline__ unsigned int pack_bf16(float lo, float hi) {
  unsigned int a = __float_as_uint(lo) + 0x8000u;
  unsigned int b = __float_as_uint(hi) + 0x8000u;
  return __builtin_amdgcn_perm(b, a, 0x07060302u);  // {b[31:16], a[31:16]}
}
__device__ __forceinline__ float lrelu(float v) { return v >= 0.f ? v : 0.1f * v; }

// CDNA5 async global->LDS copy (ASYNCcnt) -- no builtin with stable arity, use asm.
__device__ __forceinline__ void async_g2l_b128(unsigned lds_off, const void* gptr) {
  asm volatile("global_load_async_to_lds_b128 %0, %1, off"
               :: "v"(lds_off), "v"(gptr) : "memory");
}
__device__ __forceinline__ void wait_asynccnt0() {
  asm volatile("s_wait_asynccnt 0" ::: "memory");
}

// ---------------------------------------------------------------------------
// 1) Encoder grouped conv: x[b, g*20..] dot enc_w[g,c,:] + enc_b -> leaky -> bf16
// ---------------------------------------------------------------------------
__global__ __launch_bounds__(256) void enc_conv_kernel(
    const float* __restrict__ x, const float* __restrict__ w,
    const float* __restrict__ bias, unsigned short* __restrict__ h) {
  int g = blockIdx.x * 256 + threadIdx.x;
  int b = blockIdx.y;
  if (g >= NGRP) return;
  float xb[KW];
  const float4* xp = (const float4*)(x + (size_t)b * INLEN + (size_t)g * KW);
#pragma unroll
  for (int i = 0; i < 5; ++i) ((float4*)xb)[i] = xp[i];
  float wb[NCH * KW];
  const float4* wp = (const float4*)(w + (size_t)g * (NCH * KW));
#pragma unroll
  for (int i = 0; i < 15; ++i) ((float4*)wb)[i] = wp[i];
  unsigned short* hp = h + (size_t)b * GC + (size_t)g * NCH;
#pragma unroll
  for (int c = 0; c < NCH; ++c) {
    float acc = bias[g * NCH + c];
#pragma unroll
    for (int k = 0; k < KW; ++k) acc += xb[k] * wb[c * KW + k];
    hp[c] = f2bf(lrelu(acc));
  }
}

// ---------------------------------------------------------------------------
// WMMA GEMM: C[128,N] = A_bf16[128,K] x W_f32[N,K]^T, weights converted inline.
// mode 0: write f32 partial (K-split slice blockIdx.y) into outF[split][128][Npad]
// mode 1: fused bias + leaky-relu, write f32 outF[128][N]
// ---------------------------------------------------------------------------
__global__ __launch_bounds__(256) void gemm_bf16_wmma(
    const unsigned short* __restrict__ A, const float* __restrict__ W,
    const float* __restrict__ bias, float* __restrict__ outF,
    int N, int K, int Npad, int nsplit, int mode) {
  __shared__ __align__(16) unsigned short sA[2][TM * SSTR];
  __shared__ __align__(16) unsigned short sB[2][TN * SSTR];

  const int tid  = threadIdx.x;
  const int wave = tid >> 5;
  const int lane = tid & 31;
  const int half = lane >> 4;     // K-half selector per ISA 16-bit A layout
  const int lr   = lane & 15;     // row (A) / col (B) within 16x16 tile
  const int mw   = wave & 3;      // wave's M-pair
  const int nw   = wave >> 2;     // wave's N-pair
  const int nBase = blockIdx.x * TN;
  const int split = blockIdx.y;

  const int totalSteps = (K + KT - 1) / KT;
  const int t0 = (int)(((long long)totalSteps * split) / nsplit);
  const int t1 = (int)(((long long)totalSteps * (split + 1)) / nsplit);

  v8f acc[2][2] = {};
  float4 rw[2];

  // A tile: 128x32 bf16 = 512 x b128 chunks -> async copy straight into LDS.
  auto asyncLoadA = [&](int t, int buf) {
#pragma unroll
    for (int i = 0; i < 2; ++i) {
      int c = tid + i * 256;
      int r = c >> 2, j = c & 3;
      int kg = t * KT + j * 8;
      unsigned short* ldsp = &sA[buf][r * SSTR + j * 8];
      if (kg < K) {            // K%8==0 -> chunk fully valid or fully out
        async_g2l_b128((unsigned)(unsigned long long)ldsp,
                       A + (size_t)r * K + kg);
      } else {
        *(uint4*)ldsp = (uint4){0u, 0u, 0u, 0u};
      }
    }
  };

  // W tile: 64x32 f32 = 512 x float4 chunks -> regs (converted at stage time).
  auto loadW = [&](int t) {
#pragma unroll
    for (int i = 0; i < 2; ++i) {
      int c = tid + i * 256;
      int n = c >> 3, j = c & 7;
      int kg = t * KT + j * 4;
      int ng = nBase + n;
      float4 v = {0.f, 0.f, 0.f, 0.f};
      if (ng < N && kg < K) {
        v = *(const float4*)(W + (size_t)ng * K + kg);
        if (kg + 2 * KT < K)                // stream weights ahead into GL2
          __builtin_prefetch(W + (size_t)ng * K + kg + 2 * KT, 0, 0);
      }
      rw[i] = v;
    }
  };

  auto stageW = [&](int buf) {
#pragma unroll
    for (int i = 0; i < 2; ++i) {           // f32 -> bf16: 1 add/float + 1 perm/pair
      int c = tid + i * 256;
      int n = c >> 3, j = c & 7;
      uint2 pk;
      pk.x = pack_bf16(rw[i].x, rw[i].y);
      pk.y = pack_bf16(rw[i].z, rw[i].w);
      *(uint2*)&sB[buf][n * SSTR + j * 4] = pk;
    }
  };

  auto computeStep = [&](int buf) {
    Frag16 fa[2], fb[2];
    // ISA 7.12.2 16-bit A 16x32: lane lr = row, halves hold K {0..7,16..23} / {8..15,24..31}
#pragma unroll
    for (int i = 0; i < 2; ++i) {
      const unsigned short* p = &sA[buf][(32 * mw + 16 * i + lr) * SSTR + 8 * half];
      fa[i].q[0] = *(const uint4*)p;
      fa[i].q[1] = *(const uint4*)(p + 16);
    }
#pragma unroll
    for (int j = 0; j < 2; ++j) {           // B mirrors A with N as lane index
      const unsigned short* p = &sB[buf][(32 * nw + 16 * j + lr) * SSTR + 8 * half];
      fb[j].q[0] = *(const uint4*)p;
      fb[j].q[1] = *(const uint4*)(p + 16);
    }
#pragma unroll
    for (int i = 0; i < 2; ++i)
#pragma unroll
      for (int j = 0; j < 2; ++j)
        acc[i][j] = __builtin_amdgcn_wmma_f32_16x16x32_bf16(
            false, fa[i].v, false, fb[j].v, (short)0, acc[i][j], false, false);
  };

  // Prologue: fill buffer 0.
  asyncLoadA(t0, 0);
  loadW(t0);
  stageW(0);
  wait_asynccnt0();
  __syncthreads();

  // Single-barrier double-buffered pipeline:
  //   issue async A(t+1)->nbuf, load W(t+1) | compute(buf) | stage W->nbuf,
  //   wait own asynccnt | barrier publishes nbuf.
  int buf = 0;
  for (int t = t0; t < t1; ++t) {
    bool more = (t + 1) < t1;
    if (more) {
      asyncLoadA(t + 1, buf ^ 1);
      loadW(t + 1);
    }
    computeStep(buf);
    if (more) {
      stageW(buf ^ 1);
      wait_asynccnt0();
    }
    __syncthreads();
    buf ^= 1;
  }

  // C layout: VGPR e -> M = e + 8*half, N = lane&15
#pragma unroll
  for (int i = 0; i < 2; ++i) {
#pragma unroll
    for (int j = 0; j < 2; ++j) {
      int nCol = nBase + 32 * nw + 16 * j + lr;
      if (nCol < N) {
        int mBase = 32 * mw + 16 * i + 8 * half;
        if (mode == 0) {
          float* dst = outF + ((size_t)split * TM + mBase) * (size_t)Npad + nCol;
#pragma unroll
          for (int e = 0; e < 8; ++e) dst[(size_t)e * Npad] = acc[i][j][e];
        } else {
          float bv = bias[nCol];
#pragma unroll
          for (int e = 0; e < 8; ++e) {
            float v = lrelu(acc[i][j][e] + bv);
            outF[(size_t)(mBase + e) * N + nCol] = v;
          }
        }
      }
    }
  }
}

// ---------------------------------------------------------------------------
// 3) K-split reduction + bias + leaky -> z bf16
// ---------------------------------------------------------------------------
__global__ __launch_bounds__(256) void reduce_bias_kernel(
    const float* __restrict__ p, const float* __restrict__ bias,
    unsigned short* __restrict__ z, int M, int N, int Npad, int nsplit) {
  int idx = blockIdx.x * 256 + threadIdx.x;
  if (idx >= M * N) return;
  int m = idx / N, n = idx - m * N;
  float s = 0.f;
  for (int sp = 0; sp < nsplit; ++sp) s += p[((size_t)sp * M + m) * (size_t)Npad + n];
  s += bias[n];
  z[idx] = f2bf(lrelu(s));
}

// ---------------------------------------------------------------------------
// 5) Decoder grouped convT + sigmoid
// ---------------------------------------------------------------------------
__global__ __launch_bounds__(256) void dec_conv_kernel(
    const float* __restrict__ d, const float* __restrict__ w,
    const float* __restrict__ bias, float* __restrict__ out) {
  int g = blockIdx.x * 256 + threadIdx.x;
  int b = blockIdx.y;
  if (g >= NGRP) return;
  const float* dp = d + (size_t)b * GC + (size_t)g * NCH;
  float d0 = dp[0], d1 = dp[1], d2 = dp[2];
  float wb[NCH * KW];
  const float4* wp = (const float4*)(w + (size_t)g * (NCH * KW));
#pragma unroll
  for (int i = 0; i < 15; ++i) ((float4*)wb)[i] = wp[i];
  float bv = bias[g];
  float ob[KW];
#pragma unroll
  for (int k = 0; k < KW; ++k) {
    float v = d0 * wb[k] + d1 * wb[KW + k] + d2 * wb[2 * KW + k] + bv;
    ob[k] = 1.0f / (1.0f + __expf(-v));
  }
  float4* op = (float4*)(out + (size_t)b * INLEN + (size_t)g * KW);
#pragma unroll
  for (int i = 0; i < 5; ++i) op[i] = ((float4*)ob)[i];
}

// ---------------------------------------------------------------------------
extern "C" void kernel_launch(void* const* d_in, const int* in_sizes, int n_in,
                              void* d_out, int out_size, void* d_ws, size_t ws_size,
                              hipStream_t stream) {
  (void)in_sizes; (void)n_in; (void)out_size; (void)ws_size;
  const float* x        = (const float*)d_in[0];
  const float* enc_w    = (const float*)d_in[1];
  const float* enc_b    = (const float*)d_in[2];
  const float* enc_fc_w = (const float*)d_in[3];
  const float* enc_fc_b = (const float*)d_in[4];
  const float* dec_fc_w = (const float*)d_in[5];
  const float* dec_fc_b = (const float*)d_in[6];
  const float* dec_w    = (const float*)d_in[7];
  const float* dec_b    = (const float*)d_in[8];
  float* out = (float*)d_out;

  char* ws = (char*)d_ws;
  unsigned short* h_bf   = (unsigned short*)(ws);              //  7,680,000 B
  unsigned short* z_bf   = (unsigned short*)(ws +  7680000);   //    512,000 B
  float*          d_f32  = (float*)(ws +  8192000);            // 15,360,000 B
  float*          part   = (float*)(ws + 23552000);            //  8,388,608 B

  enc_conv_kernel<<<dim3((NGRP + 255) / 256, BATCH), 256, 0, stream>>>(
      x, enc_w, enc_b, h_bf);

  gemm_bf16_wmma<<<dim3((LAT + TN - 1) / TN, NSPLIT), 256, 0, stream>>>(
      h_bf, enc_fc_w, nullptr, part, LAT, GC, NPAD, NSPLIT, 0);

  reduce_bias_kernel<<<(BATCH * LAT + 255) / 256, 256, 0, stream>>>(
      part, enc_fc_b, z_bf, BATCH, LAT, NPAD, NSPLIT);

  gemm_bf16_wmma<<<dim3((GC + TN - 1) / TN, 1), 256, 0, stream>>>(
      z_bf, dec_fc_w, dec_fc_b, d_f32, GC, LAT, GC, 1, 1);

  dec_conv_kernel<<<dim3((NGRP + 255) / 256, BATCH), 256, 0, stream>>>(
      d_f32, dec_w, dec_b, out);
}
